// CRvNN2_32341103738886
// MI455X (gfx1250) — compile-verified, matrix operand order checked by reading, not run on hardware
//
#include <hip/hip_runtime.h>
#include <hip/hip_bf16.h>

typedef __attribute__((ext_vector_type(16))) _Float16 v16h;
typedef __attribute__((ext_vector_type(8)))  _Float16 v8h;
typedef __attribute__((ext_vector_type(4)))  _Float16 f16x4;
typedef __attribute__((ext_vector_type(4)))  float    f32x4;
typedef __attribute__((ext_vector_type(8)))  float    v8f;

#define EPSV 1e-9f
#define LOG2E 1.4426950408889634f

// branch-free hardware transcendentals (v_exp_f32 / v_rcp_f32, TRANS unit)
__device__ __forceinline__ float fast_exp(float x) {
  return __builtin_amdgcn_exp2f(x * LOG2E);
}
__device__ __forceinline__ float fast_rcp(float x) {
  return __builtin_amdgcn_rcpf(x);
}
__device__ __forceinline__ float fast_tanh(float x) {
  // tanh(x) = 1 - 2/(1+exp(2x)); exp2 saturates -> correct +-1 tails, no branches
  float e = __builtin_amdgcn_exp2f(x * (2.0f * LOG2E));
  return 1.0f - 2.0f * fast_rcp(e + 1.0f);
}
__device__ __forceinline__ float gelu_tanh(float x) {
  float x3 = x * x * x;
  return 0.5f * x * (1.0f + fast_tanh(0.7978845608028654f * (x + 0.044715f * x3)));
}
__device__ __forceinline__ float sigmoidf_fast(float x) {
  return fast_rcp(1.0f + __builtin_amdgcn_exp2f(-x * LOG2E));
}

// order-preserving float<->uint key for atomicMax over signed floats
__device__ __forceinline__ unsigned fkey(float f) {
  unsigned u = __float_as_uint(f);
  return (u & 0x80000000u) ? ~u : (u | 0x80000000u);
}
__device__ __forceinline__ float funkey(unsigned k) {
  return (k & 0x80000000u) ? __uint_as_float(k & 0x7fffffffu) : __uint_as_float(~k);
}

// ---------------------------------------------------------------------------
// WMMA GEMM: C[M,Ncols] = act(A[M,K] @ B[K,Ncols] + bias)
// Requirements (hold for every call in this model): K % 32 == 0, Ncols % 128 == 0.
// Block 256 thr = 8 waves; block tile 128x128; wave tile 32x64
// (2x4 v_wmma_f32_16x16x32_f16 per 32-wide K chunk).
// A rows are clamped for loads (always-valid addresses); only stores check M.
// ---------------------------------------------------------------------------
__global__ __launch_bounds__(256) void wmma_gemm_kernel(
    const float* __restrict__ A, const float* __restrict__ B,
    const float* __restrict__ bias, float* __restrict__ C,
    int M, int Ncols, int K, int act)
{
  __shared__ _Float16 As[128][32];  // [m][k]
  __shared__ _Float16 Bs[128][32];  // transposed: [n][k]

  const int tid  = threadIdx.x;
  const int wave = tid >> 5;
  const int lane = tid & 31;
  const int wm = wave & 3;          // 4 waves along M
  const int wn = wave >> 2;         // 2 waves along N
  const int laneHi = lane >> 4;     // 0: lanes 0-15, 1: lanes 16-31
  const int ln = lane & 15;
  const long bM = (long)blockIdx.y * 128;
  const long bN = (long)blockIdx.x * 128;

  v8f acc[2][4] = {{{}, {}, {}, {}}, {{}, {}, {}, {}}};

  for (int k0 = 0; k0 < K; k0 += 32) {
    // --- stage A tile 128x32: one float4 per thread per pass, b128 loads ---
    #pragma unroll
    for (int pass = 0; pass < 4; ++pass) {
      int idx = pass * 256 + tid;
      int r  = idx >> 3;            // 0..127
      int c4 = (idx & 7) << 2;      // 0,4,...,28
      long gm = bM + r; if (gm >= M) gm = M - 1;   // clamp: valid addr, row unused
      f32x4 v = *(const f32x4*)(A + gm * (long)K + (k0 + c4));
      *(f16x4*)(&As[r][c4]) = __builtin_convertvector(v, f16x4);
    }
    // --- stage B tile 32x128 -> Bs[n][k]: 4 k-values per thread per pass ---
    #pragma unroll
    for (int pass = 0; pass < 4; ++pass) {
      int idx = pass * 256 + tid;
      int n  = idx & 127;           // coalesced across lanes
      int k4 = (idx >> 7) << 2;     // 0,4,...,28
      const float* bp = B + (long)(k0 + k4) * Ncols + bN + n;
      f16x4 h;
      h.x = (_Float16)bp[0];
      h.y = (_Float16)bp[(long)Ncols];
      h.z = (_Float16)bp[2L * Ncols];
      h.w = (_Float16)bp[3L * Ncols];
      *(f16x4*)(&Bs[n][k4]) = h;    // single ds_store_b64
    }
    __syncthreads();

    // 16-bit A fragment layout (ISA 7.12.2): lanes 0-15 row M=lane, halves
    // e0..7 -> K 0..7, e8..15 -> K 16..23; lanes 16-31 -> K+8. Two contiguous
    // v8h LDS reads per fragment. B mirrors with lane = N.
    v16h af[2], bf[4];
    #pragma unroll
    for (int mi = 0; mi < 2; ++mi) {
      const _Float16* p = &As[wm * 32 + mi * 16 + ln][0];
      v8h lo = *(const v8h*)(p + laneHi * 8);
      v8h hi = *(const v8h*)(p + 16 + laneHi * 8);
      af[mi] = __builtin_shufflevector(lo, hi, 0,1,2,3,4,5,6,7,8,9,10,11,12,13,14,15);
    }
    #pragma unroll
    for (int ni = 0; ni < 4; ++ni) {
      const _Float16* p = &Bs[wn * 64 + ni * 16 + ln][0];
      v8h lo = *(const v8h*)(p + laneHi * 8);
      v8h hi = *(const v8h*)(p + 16 + laneHi * 8);
      bf[ni] = __builtin_shufflevector(lo, hi, 0,1,2,3,4,5,6,7,8,9,10,11,12,13,14,15);
    }
    #pragma unroll
    for (int mi = 0; mi < 2; ++mi)
      #pragma unroll
      for (int ni = 0; ni < 4; ++ni)
        acc[mi][ni] = __builtin_amdgcn_wmma_f32_16x16x32_f16(
            false, af[mi], false, bf[ni], (short)0, acc[mi][ni], false, false);
    __syncthreads();
  }

  // C/D layout: VGPR r -> M = r (lanes 0-15) / r+8 (lanes 16-31), N = lane%16
  #pragma unroll
  for (int mi = 0; mi < 2; ++mi) {
    #pragma unroll
    for (int ni = 0; ni < 4; ++ni) {
      long cm = bM + wm * 32 + mi * 16 + laneHi * 8;
      long cn = bN + wn * 64 + ni * 16 + ln;      // always < Ncols (Ncols%128==0)
      float bb = bias ? bias[cn] : 0.0f;
      #pragma unroll
      for (int r = 0; r < 8; ++r) {
        long gm = cm + r;
        if (gm < M) {
          float v = acc[mi][ni][r] + bb;
          if (act == 1) v = gelu_tanh(v);
          C[gm * (long)Ncols + cn] = v;
        }
      }
    }
  }
}

// ---------------------------------------------------------------------------
// Build augmented sequence + masks. One block per row (n,i), blockDim = D.
// ---------------------------------------------------------------------------
__global__ void init0_kernel(const float* __restrict__ seqin, const float* __restrict__ inmask,
                             const float* __restrict__ START, const float* __restrict__ END,
                             float* __restrict__ seq0, float* __restrict__ mask,
                             float* __restrict__ selp, float* __restrict__ active,
                             float* __restrict__ ltp, int S, int D)
{
  const int S2 = S + 2;
  const long row = blockIdx.x;
  const int n = (int)(row / S2), i = (int)(row % S2);

  // m1[j] (len S+1): j==0 -> 1 else input_mask[n, j-1]
  float m1_i   = (i == 0) ? 1.0f : ((i <= S) ? inmask[(long)n * S + (i - 1)] : 0.0f);
  float m1_im1 = (i - 1 == 0) ? 1.0f : ((i - 1 >= 1 && i - 1 <= S) ? inmask[(long)n * S + (i - 2)] : 0.0f);
  float maskNoEnd  = (i <= S) ? m1_i : 0.0f;
  float maskYesEnd = (i == 0) ? 1.0f : m1_im1;
  float endm = maskYesEnd - maskNoEnd;
  float mk = maskYesEnd;
  float maskNoStart = (i == 0) ? 0.0f : mk;
  float ltm = 0.0f;
  if (i < S2 - 1) {
    int i2 = i + 1;
    float m1_i2   = (i2 <= S) ? inmask[(long)n * S + (i2 - 1)] : 0.0f;
    float m1_i2m1 = (i2 - 1 == 0) ? 1.0f : ((i2 - 1 <= S) ? inmask[(long)n * S + (i2 - 2)] : 0.0f);
    float mne2 = (i2 <= S) ? m1_i2 : 0.0f;
    float mye2 = m1_i2m1;
    ltm = mye2 - mne2;
  }
  float sp = maskNoStart * maskNoEnd * (1.0f - ltm);

  for (int d = threadIdx.x; d < D; d += blockDim.x) {
    float se;
    if (i == 0)      se = START[d];
    else if (i <= S) se = seqin[((long)n * S + (i - 1)) * D + d];
    else             se = 0.0f;
    seq0[row * (long)D + d] = endm * END[d] + (1.0f - endm) * se;
  }
  if (threadIdx.x == 0) {
    mask[row] = mk; selp[row] = sp; active[row] = mk; ltp[row] = 0.0f;
  }
}

// LayerNorm * mask, block per row, blockDim == D == 256
__global__ void ln_mask_kernel(const float* __restrict__ x, const float* __restrict__ g,
                               const float* __restrict__ b, const float* __restrict__ mask,
                               float* __restrict__ out, int D)
{
  __shared__ float red[256];
  const long row = blockIdx.x;
  const int d = threadIdx.x;
  float v = x[row * (long)D + d];
  red[d] = v; __syncthreads();
  for (int s = 128; s > 0; s >>= 1) { if (d < s) red[d] += red[d + s]; __syncthreads(); }
  float m = red[0] / D; __syncthreads();
  float c = v - m;
  red[d] = c * c; __syncthreads();
  for (int s = 128; s > 0; s >>= 1) { if (d < s) red[d] += red[d + s]; __syncthreads(); }
  float var = red[0] / D;
  out[row * (long)D + d] = (c * rsqrtf(var + 1e-5f) * g[d] + b[d]) * mask[row];
}

// scan coefficients q = 1 - active*mask + EPS ; w = active*mask^2
__global__ void coeffs_kernel(const float* __restrict__ active, const float* __restrict__ mask,
                              float* __restrict__ q, float* __restrict__ w, long R)
{
  long idx = (long)blockIdx.x * blockDim.x + threadIdx.x;
  if (idx < R) {
    float a = active[idx] * mask[idx];
    q[idx] = 1.0f - a + EPSV;
    w[idx] = a * mask[idx];
  }
}

// base = seq + ltp*yes + (1-ltp)*no -> window slot 2
__global__ void make_base_kernel(const float* __restrict__ seq, const float* __restrict__ ltp,
                                 const float* __restrict__ yes, const float* __restrict__ no,
                                 float* __restrict__ win, int D)
{
  const long row = blockIdx.x;
  const int d = threadIdx.x;
  float l = ltp[row];
  win[row * (5L * D) + 2L * D + d] = seq[row * (long)D + d] + l * yes[d] + (1.0f - l) * no[d];
}

// linear-recurrence realization of lnp@x (dir=0) / rnp@x (dir=1)
// out[i] = state before absorbing x[i]; state' = q[i]*state + w[i]*x[i]
__global__ void scan_kernel(const float* __restrict__ x, float* __restrict__ out,
                            const float* __restrict__ q, const float* __restrict__ w,
                            int S2, int D, int xPitch, int outPitch, int dir)
{
  const int n = blockIdx.x;
  const int d = threadIdx.x;
  const float* qn = q + (long)n * S2;
  const float* wn = w + (long)n * S2;
  const float* xn = x + (long)n * S2 * (long)xPitch;
  float* on = out + (long)n * S2 * (long)outPitch;
  float acc = 0.0f;
  if (dir == 0) {
    for (int i = 0; i < S2; ++i) {
      on[(long)i * outPitch + d] = acc;
      acc = qn[i] * acc + wn[i] * xn[(long)i * xPitch + d];
    }
  } else {
    for (int i = S2 - 1; i >= 0; --i) {
      on[(long)i * outPitch + d] = acc;
      acc = qn[i] * acc + wn[i] * xn[(long)i * xPitch + d];
    }
  }
}

__global__ void initmx_kernel(unsigned* mxkey) { *mxkey = fkey(0.0f); }

// tsc = conv_row . scW + scb ; global atomic max of tsc (clamped at 0 by init)
__global__ void score_kernel(const float* __restrict__ conv, const float* __restrict__ scW,
                             const float* __restrict__ scb, float* __restrict__ tsc,
                             unsigned* __restrict__ mxkey, int D)
{
  __shared__ float red[256];
  const long row = blockIdx.x;
  const int d = threadIdx.x;
  red[d] = conv[row * (long)D + d] * scW[d];
  __syncthreads();
  for (int s = 128; s > 0; s >>= 1) { if (d < s) red[d] += red[d + s]; __syncthreads(); }
  if (d == 0) {
    float t = red[0] + scb[0];   // TEMP == 1
    tsc[row] = t;
    atomicMax(mxkey, fkey(t));
  }
}

__global__ void tp_kernel(const float* __restrict__ tsc, const float* __restrict__ selp,
                          const unsigned* __restrict__ mxkey, float* __restrict__ tp, long R)
{
  long idx = (long)blockIdx.x * blockDim.x + threadIdx.x;
  if (idx < R) {
    float mx = funkey(*mxkey);
    float et = fast_exp(tsc[idx] - mx) * selp[idx];
    float en = fast_exp(-mx);
    tp[idx] = et * fast_rcp(et + en + EPSV);
  }
}

__global__ void pack_cc_kernel(const float* __restrict__ lc, const float* __restrict__ seq,
                               float* __restrict__ cc, int D)
{
  const long row = blockIdx.x;
  const int d = threadIdx.x;
  cc[row * 2L * D + d]     = lc[row * (long)D + d];
  cc[row * 2L * D + D + d] = seq[row * (long)D + d];
}

// gates/comp/LN/blend; block per row, blockDim == D == 256
__global__ void update_kernel(const float* __restrict__ contents, const float* __restrict__ lc,
                              float* __restrict__ seq, const float* __restrict__ tp,
                              const float* __restrict__ mask, const float* __restrict__ g,
                              const float* __restrict__ b, int D)
{
  __shared__ float red[256];
  const long row = blockIdx.x;
  const int d = threadIdx.x;
  const long cbase = row * 4L * D;
  float c0 = contents[cbase + d];
  float c1 = contents[cbase + D + d];
  float c2 = contents[cbase + 2L * D + d];
  float c3 = contents[cbase + 3L * D + d];
  float l  = lc[row * (long)D + d];
  float s0 = seq[row * (long)D + d];
  float pre = sigmoidf_fast(c0) * l + sigmoidf_fast(c1) * s0 + sigmoidf_fast(c2) * c3;
  red[d] = pre; __syncthreads();
  for (int s = 128; s > 0; s >>= 1) { if (d < s) red[d] += red[d + s]; __syncthreads(); }
  float m = red[0] / D; __syncthreads();
  float cdev = pre - m;
  red[d] = cdev * cdev; __syncthreads();
  for (int s = 128; s > 0; s >>= 1) { if (d < s) red[d] += red[d + s]; __syncthreads(); }
  float var = red[0] / D;
  float comp = cdev * rsqrtf(var + 1e-5f) * g[d] + b[d];
  float t = tp[row];
  seq[row * (long)D + d] = (t * comp + (1.0f - t) * s0) * mask[row];
}

// deact[j] = w[j] * sum_{i>j} (prod_{j<k<i} q[k]) tp[i]  (reverse scan, 1 thread/batch)
// then active = clip(active*(1-deact),0,1)*mask ; ltp = tp
__global__ void deact_kernel(const float* __restrict__ tp, const float* __restrict__ q,
                             const float* __restrict__ w, float* __restrict__ active,
                             const float* __restrict__ mask, float* __restrict__ ltp, int S2)
{
  const int n = threadIdx.x;
  const long base = (long)n * S2;
  float s = 0.0f;
  for (int j = S2 - 1; j >= 0; --j) {
    float de = w[base + j] * s;
    float a = active[base + j] * (1.0f - de);
    a = fminf(fmaxf(a, 0.0f), 1.0f) * mask[base + j];
    active[base + j] = a;
    ltp[base + j] = tp[base + j];
    s = tp[base + j] + q[base + j] * s;
  }
}

// ---------------------------------------------------------------------------
extern "C" void kernel_launch(void* const* d_in, const int* in_sizes, int n_in,
                              void* d_out, int out_size, void* d_ws, size_t ws_size,
                              hipStream_t stream)
{
  (void)in_sizes; (void)n_in; (void)out_size; (void)ws_size;
  const float* sequence = (const float*)d_in[0];
  const float* input_mask = (const float*)d_in[1];
  const float* START = (const float*)d_in[2];
  const float* END   = (const float*)d_in[3];
  const float* yes_t = (const float*)d_in[4];
  const float* no_t  = (const float*)d_in[5];
  const float* convW = (const float*)d_in[6];
  const float* convb = (const float*)d_in[7];
  const float* scW   = (const float*)d_in[8];
  const float* scb   = (const float*)d_in[9];
  const float* itW   = (const float*)d_in[10];
  const float* itb   = (const float*)d_in[11];
  const float* w1W   = (const float*)d_in[12];
  const float* w1b   = (const float*)d_in[13];
  const float* w2W   = (const float*)d_in[14];
  const float* w2b   = (const float*)d_in[15];
  const float* lng   = (const float*)d_in[16];
  const float* lnb   = (const float*)d_in[17];

  const int N = 32, S = 512, D = 256, H = 1024, S2 = S + 2, n_steps = 6;
  const long R = (long)N * S2;                 // 16448 rows

  // workspace layout (floats); inter aliases win (free after conv GEMM)
  float* p = (float*)d_ws;
  float* seq      = p; p += R * D;             // (N,S2,D)
  float* win      = p; p += R * 5L * D;        // (N,S2,5,D) == conv GEMM input
  float* convout  = p; p += R * D;
  float* lc       = p; p += R * D;             // left_child
  float* cc       = p; p += R * 2L * D;
  float* contents = p; p += R * 4L * D;
  float* inter    = win;                       // R*4D <= R*5D
  float* mask   = p; p += R;
  float* selp   = p; p += R;
  float* active = p; p += R;
  float* ltp    = p; p += R;
  float* qb     = p; p += R;
  float* wb     = p; p += R;
  float* tsc    = p; p += R;
  float* tpb    = p; p += R;
  unsigned* mxkey = (unsigned*)p;

  const dim3 blk(256);
  const int eltBlocks = (int)((R + 255) / 256);
  auto gemmGrid = [](long M, int Nc) { return dim3((unsigned)((Nc + 127) / 128), (unsigned)((M + 127) / 128)); };

  // ---- init: seq_ext/masks -> seq = LN(seq_ext @ itW + itb) * mask
  init0_kernel<<<(unsigned)R, blk, 0, stream>>>(sequence, input_mask, START, END,
                                                win, mask, selp, active, ltp, S, D);
  wmma_gemm_kernel<<<gemmGrid(R, D), blk, 0, stream>>>(win, itW, itb, convout,
                                                       (int)R, D, D, 0);
  ln_mask_kernel<<<(unsigned)R, blk, 0, stream>>>(convout, lng, lnb, mask, seq, D);

  for (int step = 0; step < n_steps; ++step) {
    coeffs_kernel<<<eltBlocks, blk, 0, stream>>>(active, mask, qb, wb, R);
    make_base_kernel<<<(unsigned)R, blk, 0, stream>>>(seq, ltp, yes_t, no_t, win, D);

    // window: [L2, L1, base, R1, R2] via left/right linear-recurrence scans
    scan_kernel<<<N, blk, 0, stream>>>(win + 2L * D, win + 1L * D, qb, wb, S2, D, 5 * D, 5 * D, 0);
    scan_kernel<<<N, blk, 0, stream>>>(win + 1L * D, win,          qb, wb, S2, D, 5 * D, 5 * D, 0);
    scan_kernel<<<N, blk, 0, stream>>>(win + 2L * D, win + 3L * D, qb, wb, S2, D, 5 * D, 5 * D, 1);
    scan_kernel<<<N, blk, 0, stream>>>(win + 3L * D, win + 4L * D, qb, wb, S2, D, 5 * D, 5 * D, 1);

    // conv: gelu(flat @ convW + convb)
    wmma_gemm_kernel<<<gemmGrid(R, D), blk, 0, stream>>>(win, convW, convb, convout,
                                                         (int)R, D, 5 * D, 1);
    initmx_kernel<<<1, 1, 0, stream>>>(mxkey);
    score_kernel<<<(unsigned)R, blk, 0, stream>>>(convout, scW, scb, tsc, mxkey, D);
    tp_kernel<<<eltBlocks, blk, 0, stream>>>(tsc, selp, mxkey, tpb, R);

    // left_child = lnp @ seq (scan), then FFN GEMMs
    scan_kernel<<<N, blk, 0, stream>>>(seq, lc, qb, wb, S2, D, D, D, 0);
    pack_cc_kernel<<<(unsigned)R, blk, 0, stream>>>(lc, seq, cc, D);
    wmma_gemm_kernel<<<gemmGrid(R, H), blk, 0, stream>>>(cc, w1W, w1b, inter,
                                                         (int)R, H, 2 * D, 1);
    wmma_gemm_kernel<<<gemmGrid(R, 4 * D), blk, 0, stream>>>(inter, w2W, w2b, contents,
                                                             (int)R, 4 * D, H, 0);

    update_kernel<<<(unsigned)R, blk, 0, stream>>>(contents, lc, seq, tpb, mask, lng, lnb, D);
    deact_kernel<<<1, N, 0, stream>>>(tpb, qb, wb, active, mask, ltp, S2);
  }

  hipMemcpyAsync(d_out, seq, (size_t)R * D * sizeof(float),
                 hipMemcpyDeviceToDevice, stream);
}
